// MoETransformer_24584392802846
// MI455X (gfx1250) — matrix operand order repaired and checked by reading, not compiled, not run on hardware
//
#include <hip/hip_runtime.h>
#include <hip/hip_bf16.h>
#include <math.h>

// ---- problem dims (match reference) ----
#define B_   4
#define S_   512
#define H_   512
#define NH_  8
#define HD_  64
#define E_   8
#define KTOP 2
#define L_   6
#define V_   50000
#define F_   2048
#define T_   (B_*S_)    // 2048 tokens
#define EPS_ 1e-5f

typedef __attribute__((ext_vector_type(16))) __bf16 v16bf;
typedef __attribute__((ext_vector_type(8)))  __bf16 v8bf;
typedef __attribute__((ext_vector_type(8)))  float  v8f;

__device__ __forceinline__ __bf16 f2bf(float f)  { return (__bf16)f; }
__device__ __forceinline__ float  bf2f(__bf16 h) { return (float)h; }

__device__ __forceinline__ float gelu_exact(float v) {
  return 0.5f * v * (1.0f + erff(v * 0.70710678118654752440f));
}

__device__ __forceinline__ v16bf load_afrag(const __bf16* arow, int k0, int half) {
  v8bf alo = *(const v8bf*)(arow + k0 + half * 16);
  v8bf ahi = *(const v8bf*)(arow + k0 + half * 16 + 8);
  return __builtin_shufflevector(alo, ahi,
      0, 1, 2, 3, 4, 5, 6, 7, 8, 9, 10, 11, 12, 13, 14, 15);
}

// =====================================================================
// NT WMMA GEMM, all-bf16 operands:
//   C[z,y](M,N) = scale * A(M,K,row-major) * B(N,K,row-major)^T + bias
// Wave computes 32x64 (2 A-frags x 4 N-subtiles = 8 acc): each B
// fragment is reused by two WMMAs -> 12 b128 loads per 8 WMMA issues.
// Fragment layouts used here (wave32; A/B k order is contiguous, not ISA table order):
//   A 16x32: lane(m=lane&15): K = half*16..+15
//   B 32x16: lane(n=lane&15): K = half*16..+15
//   C 16x16: M = r + half*8, N = lane&15
// =====================================================================
template <int OUTBF16, int HASBIAS>
__global__ void gemm_nt(const __bf16* __restrict__ A, int lda, long long sAy, long long sAz,
                        const __bf16* __restrict__ Bm, int ldb, long long sBy, long long sBz,
                        const float* __restrict__ bias,
                        void* __restrict__ Cv, int ldc, long long sCy, long long sCz,
                        int M, int N, int K, float scale) {
  A  += (long long)blockIdx.y * sAy + (long long)blockIdx.z * sAz;
  Bm += (long long)blockIdx.y * sBy + (long long)blockIdx.z * sBz;

  const int lane = threadIdx.x & 31;
  const int wid  = blockIdx.x * 8 + (threadIdx.x >> 5);
  const int tilesN = (N + 63) >> 6;
  const int tM = wid / tilesN, tN = wid % tilesN;
  if (tM * 32 >= M) return;

  const int half  = lane >> 4;
  const int nbase = tN * 64;
  const int ncol  = lane & 15;
  const __bf16* arow0 = A + (size_t)(tM * 32 + (lane & 15)) * lda;
  const __bf16* arow1 = arow0 + (size_t)16 * lda;
  const __bf16* b0p = Bm + (size_t)(nbase + ncol) * ldb + half * 16;

  v8f acc[2][4] = {};
  for (int k0 = 0; k0 < K; k0 += 32) {
    v16bf a0 = load_afrag(arow0, k0, half);
    v16bf a1 = load_afrag(arow1, k0, half);
    __builtin_prefetch(arow0 + k0 + 128, 0, 1);  // global_prefetch_b8
#pragma unroll
    for (int j = 0; j < 4; ++j) {
      int n0 = nbase + j * 16;
      if (n0 >= N) break;
      // B fragment: one contiguous 32B run per lane, reused twice
      v16bf b = *(const v16bf*)(b0p + (size_t)j * 16 * ldb + k0);
      acc[0][j] = __builtin_amdgcn_wmma_f32_16x16x32_bf16(false, a0, false, b,
                                                          (short)0, acc[0][j], false, false);
      acc[1][j] = __builtin_amdgcn_wmma_f32_16x16x32_bf16(false, a1, false, b,
                                                          (short)0, acc[1][j], false, false);
    }
  }

#pragma unroll
  for (int f = 0; f < 2; ++f) {
    const int mb = tM * 32 + f * 16 + half * 8;
#pragma unroll
    for (int j = 0; j < 4; ++j) {
      int n0 = nbase + j * 16;
      if (n0 >= N) break;
      int n = n0 + ncol;
      float bv = HASBIAS ? bias[n] : 0.0f;
#pragma unroll
      for (int r = 0; r < 8; ++r) {
        float v = acc[f][j][r] * scale + bv;
        size_t idx = (size_t)(mb + r) * ldc + n
                   + (size_t)blockIdx.y * sCy + (size_t)blockIdx.z * sCz;
        if (OUTBF16) ((__bf16*)Cv)[idx] = f2bf(v);
        else         ((float*)Cv)[idx] = v;
      }
    }
  }
}

// =====================================================================
// Expert fc1: A = gathered+normalized bf16 rows (yb), B = w1T[e] (F,H)
// epilogue: bias + exact GELU -> bf16 hbuf. 32x64 wave tiles.
// =====================================================================
__global__ void moe_fc1(const __bf16* __restrict__ yb,
                        const __bf16* __restrict__ w1T, const float* __restrict__ b1,
                        const int* __restrict__ cnt, const int* __restrict__ basep,
                        __bf16* __restrict__ hbuf) {
  const int e = blockIdx.y;
  const int nc = cnt[e];
  const int lane = threadIdx.x & 31;
  const int wid = blockIdx.x * 8 + (threadIdx.x >> 5);
  const int tilesN = F_ / 64;  // 32
  const int tM = wid / tilesN, tN = wid % tilesN;
  if (tM * 32 >= nc) return;

  const int half = lane >> 4;
  const int base = basep[e];
  int p0 = tM * 32 + (lane & 15);
  int p1 = p0 + 16;
  int pc0 = (p0 < nc) ? p0 : (nc - 1);   // clamp reads into compact buffer
  int pc1 = (p1 < nc) ? p1 : (nc - 1);
  const __bf16* arow0 = yb + (size_t)(base + pc0) * H_;
  const __bf16* arow1 = yb + (size_t)(base + pc1) * H_;
  const __bf16* W = w1T + (size_t)e * F_ * H_;
  const float* brow = b1 + (size_t)e * F_;
  const int nbase = tN * 64, ncol = lane & 15;
  const __bf16* b0p = W + (size_t)(nbase + ncol) * H_ + half * 16;

  v8f acc[2][4] = {};
  for (int k0 = 0; k0 < H_; k0 += 32) {
    v16bf a0 = load_afrag(arow0, k0, half);
    v16bf a1 = load_afrag(arow1, k0, half);
#pragma unroll
    for (int j = 0; j < 4; ++j) {
      v16bf b = *(const v16bf*)(b0p + (size_t)j * 16 * H_ + k0);
      acc[0][j] = __builtin_amdgcn_wmma_f32_16x16x32_bf16(false, a0, false, b,
                                                          (short)0, acc[0][j], false, false);
      acc[1][j] = __builtin_amdgcn_wmma_f32_16x16x32_bf16(false, a1, false, b,
                                                          (short)0, acc[1][j], false, false);
    }
  }

#pragma unroll
  for (int f = 0; f < 2; ++f) {
    const int mb = tM * 32 + f * 16 + half * 8;
#pragma unroll
    for (int j = 0; j < 4; ++j) {
      int n = nbase + j * 16 + ncol;
      float bv = brow[n];
#pragma unroll
      for (int r8 = 0; r8 < 8; ++r8) {
        int pm = mb + r8;
        if (pm < nc)
          hbuf[(size_t)(base + pm) * F_ + n] = f2bf(gelu_exact(acc[f][j][r8] + bv));
      }
    }
  }
}

// =====================================================================
// Expert fc2: A = hbuf bf16, B = w2T[e] (H,F); epilogue: +b2, * route
// weight, scatter to contrib[slot][token]  (conflict-free, deterministic)
// =====================================================================
__global__ void moe_fc2(const __bf16* __restrict__ hbuf,
                        const __bf16* __restrict__ w2T, const float* __restrict__ b2,
                        const int* __restrict__ cnt, const int* __restrict__ basep,
                        const int* __restrict__ tok, const float* __restrict__ wgt,
                        const int* __restrict__ slot, float* __restrict__ contrib) {
  const int e = blockIdx.y;
  const int nc = cnt[e];
  const int lane = threadIdx.x & 31;
  const int wid = blockIdx.x * 8 + (threadIdx.x >> 5);
  const int tilesN = H_ / 64;  // 8
  const int tM = wid / tilesN, tN = wid % tilesN;
  if (tM * 32 >= nc) return;

  const int half = lane >> 4;
  const int base = basep[e];
  int p0 = tM * 32 + (lane & 15);
  int p1 = p0 + 16;
  int pc0 = (p0 < nc) ? p0 : (nc - 1);
  int pc1 = (p1 < nc) ? p1 : (nc - 1);
  const __bf16* arow0 = hbuf + (size_t)(base + pc0) * F_;
  const __bf16* arow1 = hbuf + (size_t)(base + pc1) * F_;
  const __bf16* W = w2T + (size_t)e * H_ * F_;
  const float* brow = b2 + (size_t)e * H_;
  const int nbase = tN * 64, ncol = lane & 15;
  const __bf16* b0p = W + (size_t)(nbase + ncol) * F_ + half * 16;

  v8f acc[2][4] = {};
  for (int k0 = 0; k0 < F_; k0 += 32) {
    v16bf a0 = load_afrag(arow0, k0, half);
    v16bf a1 = load_afrag(arow1, k0, half);
#pragma unroll
    for (int j = 0; j < 4; ++j) {
      v16bf b = *(const v16bf*)(b0p + (size_t)j * 16 * F_ + k0);
      acc[0][j] = __builtin_amdgcn_wmma_f32_16x16x32_bf16(false, a0, false, b,
                                                          (short)0, acc[0][j], false, false);
      acc[1][j] = __builtin_amdgcn_wmma_f32_16x16x32_bf16(false, a1, false, b,
                                                          (short)0, acc[1][j], false, false);
    }
  }

#pragma unroll
  for (int f = 0; f < 2; ++f) {
    const int mb = tM * 32 + f * 16 + half * 8;
#pragma unroll
    for (int r8 = 0; r8 < 8; ++r8) {
      int pm = mb + r8;
      if (pm >= nc) continue;
      int tt = tok[e * T_ + pm];
      float w = wgt[e * T_ + pm];
      int s = slot[e * T_ + pm];
      float* outp = contrib + (size_t)s * T_ * H_ + (size_t)tt * H_;
#pragma unroll
      for (int j = 0; j < 4; ++j) {
        int n = nbase + j * 16 + ncol;
        outp[n] = (acc[f][j][r8] + brow[n]) * w;
      }
    }
  }
}

// =====================================================================
// Transpose+convert to bf16: dst(C,R) = src(R,C), batched (z -> z1,z2).
// 32x32 LDS tile, coalesced both sides.
// =====================================================================
template <typename TS>
__global__ void transpose_bf16(const TS* __restrict__ src, int ld_src,
                               long long sz1, long long sz2,
                               __bf16* __restrict__ dst, int ld_dst,
                               long long dz1, long long dz2,
                               int R, int C, int zdiv) {
  __shared__ __bf16 tile[32][33];
  int z1 = blockIdx.z / zdiv, z2 = blockIdx.z % zdiv;
  src += (long long)z1 * sz1 + (long long)z2 * sz2;
  dst += (long long)z1 * dz1 + (long long)z2 * dz2;
  int c0 = blockIdx.x * 32, r0 = blockIdx.y * 32;
  int tx = threadIdx.x & 31, ty = threadIdx.x >> 5;  // 32 x 8
#pragma unroll
  for (int i = 0; i < 32; i += 8) {
    int r = r0 + ty + i, c = c0 + tx;
    tile[ty + i][tx] = (r < R && c < C) ? f2bf((float)src[(size_t)r * ld_src + c])
                                        : f2bf(0.0f);
  }
  __syncthreads();
#pragma unroll
  for (int i = 0; i < 32; i += 8) {
    int r = r0 + tx, c = c0 + ty + i;
    if (r < R && c < C) dst[(size_t)c * ld_dst + r] = tile[tx][ty + i];
  }
}

// straight f32 -> bf16 convert (w_qkv, w_o already in (N,K) orientation)
__global__ void convert_bf16(const float* __restrict__ src, __bf16* __restrict__ dst,
                             int n) {
  int i = blockIdx.x * blockDim.x + threadIdx.x;
  if (i < n) dst[i] = f2bf(src[i]);
}

// ---------------- small VALU kernels ----------------
__global__ void embed_kernel(const int* __restrict__ ids, const float* __restrict__ te,
                             const float* __restrict__ pe, float* __restrict__ x,
                             __bf16* __restrict__ xb) {
  int i = blockIdx.x * blockDim.x + threadIdx.x;
  if (i >= T_ * H_) return;
  int t = i / H_, h = i - t * H_;
  int s = t % S_;
  float v = te[(size_t)ids[t] * H_ + h] + pe[(size_t)s * H_ + h];
  x[i] = v; xb[i] = f2bf(v);
}

__device__ __forceinline__ float block_red_sum(float v, float* red) {
  int tid = threadIdx.x;
  red[tid] = v; __syncthreads();
  for (int o = 128; o > 0; o >>= 1) { if (tid < o) red[tid] += red[tid + o]; __syncthreads(); }
  float s = red[0]; __syncthreads();
  return s;
}
__device__ __forceinline__ float block_red_max(float v, float* red) {
  int tid = threadIdx.x;
  red[tid] = v; __syncthreads();
  for (int o = 128; o > 0; o >>= 1) { if (tid < o) red[tid] = fmaxf(red[tid], red[tid + o]); __syncthreads(); }
  float s = red[0]; __syncthreads();
  return s;
}

// x = LN(x + a); also emits bf16 copy for downstream GEMMs
__global__ void add_ln(float* __restrict__ x, const float* __restrict__ a,
                       const float* __restrict__ g, const float* __restrict__ b,
                       __bf16* __restrict__ xb) {
  __shared__ float red[256];
  size_t row = blockIdx.x; int tid = threadIdx.x;
  float v0 = x[row * H_ + tid]       + a[row * H_ + tid];
  float v1 = x[row * H_ + tid + 256] + a[row * H_ + tid + 256];
  float m = block_red_sum(v0 + v1, red) * (1.0f / H_);
  float d0 = v0 - m, d1 = v1 - m;
  float var = block_red_sum(d0 * d0 + d1 * d1, red) * (1.0f / H_);
  float r = rsqrtf(var + EPS_);
  float o0 = d0 * r * g[tid] + b[tid];
  float o1 = d1 * r * g[tid + 256] + b[tid + 256];
  x[row * H_ + tid] = o0;        xb[row * H_ + tid] = f2bf(o0);
  x[row * H_ + tid + 256] = o1;  xb[row * H_ + tid + 256] = f2bf(o1);
}

// x = LN(2x + c0 + c1)  (MoE combine; exact residual since sum w = 1)
__global__ void combine_ln(float* __restrict__ x, const float* __restrict__ c,
                           const float* __restrict__ g, const float* __restrict__ b,
                           __bf16* __restrict__ xb) {
  __shared__ float red[256];
  size_t row = blockIdx.x; int tid = threadIdx.x;
  const float* c0 = c;
  const float* c1 = c + (size_t)T_ * H_;
  float v0 = 2.0f * x[row * H_ + tid]       + c0[row * H_ + tid]       + c1[row * H_ + tid];
  float v1 = 2.0f * x[row * H_ + tid + 256] + c0[row * H_ + tid + 256] + c1[row * H_ + tid + 256];
  float m = block_red_sum(v0 + v1, red) * (1.0f / H_);
  float d0 = v0 - m, d1 = v1 - m;
  float var = block_red_sum(d0 * d0 + d1 * d1, red) * (1.0f / H_);
  float r = rsqrtf(var + EPS_);
  float o0 = d0 * r * g[tid] + b[tid];
  float o1 = d1 * r * g[tid + 256] + b[tid + 256];
  x[row * H_ + tid] = o0;        xb[row * H_ + tid] = f2bf(o0);
  x[row * H_ + tid + 256] = o1;  xb[row * H_ + tid + 256] = f2bf(o1);
}

// per-token mean / rstd of x (expert LN shares token stats)
__global__ void row_stats(const float* __restrict__ x, float* __restrict__ mu,
                          float* __restrict__ rs) {
  __shared__ float red[256];
  size_t row = blockIdx.x; int tid = threadIdx.x;
  float v0 = x[row * H_ + tid], v1 = x[row * H_ + tid + 256];
  float m = block_red_sum(v0 + v1, red) * (1.0f / H_);
  float d0 = v0 - m, d1 = v1 - m;
  float var = block_red_sum(d0 * d0 + d1 * d1, red) * (1.0f / H_);
  if (tid == 0) { mu[row] = m; rs[row] = rsqrtf(var + EPS_); }
}

// softmax over bf16 rows of length 512 (attention scores, in place)
__global__ void softmax512(__bf16* __restrict__ sc) {
  __shared__ float red[256];
  __bf16* p = sc + (size_t)blockIdx.x * S_;
  int tid = threadIdx.x;
  float v0 = bf2f(p[tid]), v1 = bf2f(p[tid + 256]);
  float mx = block_red_max(fmaxf(v0, v1), red);
  float e0 = expf(v0 - mx), e1 = expf(v1 - mx);
  float inv = 1.0f / block_red_sum(e0 + e1, red);
  p[tid] = f2bf(e0 * inv); p[tid + 256] = f2bf(e1 * inv);
}

// gather routed token, apply per-expert LN affine, emit bf16 row
__global__ void gather_norm(const float* __restrict__ x, const float* __restrict__ mu,
                            const float* __restrict__ rs,
                            const float* __restrict__ eg, const float* __restrict__ eb,
                            const int* __restrict__ cnt, const int* __restrict__ basep,
                            const int* __restrict__ tok, __bf16* __restrict__ yb) {
  int e = blockIdx.y, p = blockIdx.x;
  if (p >= cnt[e]) return;
  int t = tok[e * T_ + p];
  const float* xr = x + (size_t)t * H_;
  float m = mu[t], r = rs[t];
  __bf16* dst = yb + (size_t)(basep[e] + p) * H_;
  for (int h = threadIdx.x; h < H_; h += 256)
    dst[h] = f2bf((xr[h] - m) * r * eg[(size_t)e * H_ + h] + eb[(size_t)e * H_ + h]);
}

// router: logits = x @ rw + rb ; top-2 ; softmax ; build per-expert lists
__global__ void router_topk(const float* __restrict__ x, const float* __restrict__ rw,
                            const float* __restrict__ rb, int* __restrict__ cnt,
                            int* __restrict__ tok, float* __restrict__ wgt,
                            int* __restrict__ slot) {
  int t = blockIdx.x * blockDim.x + threadIdx.x;
  if (t >= T_) return;
  const float* xr = x + (size_t)t * H_;
  float lg[E_];
#pragma unroll
  for (int e = 0; e < E_; ++e) lg[e] = rb[e];
  for (int h = 0; h < H_; ++h) {
    float xv = xr[h];
    const float* rwh = rw + (size_t)h * E_;
#pragma unroll
    for (int e = 0; e < E_; ++e) lg[e] += xv * rwh[e];
  }
  float b0 = -1e30f, b1 = -1e30f; int i0 = 0, i1 = 0;
#pragma unroll
  for (int e = 0; e < E_; ++e) {
    float v = lg[e];
    if (v > b0)      { b1 = b0; i1 = i0; b0 = v; i0 = e; }
    else if (v > b1) { b1 = v; i1 = e; }
  }
  float e1 = expf(b1 - b0);
  float s = 1.0f + e1;
  float p0 = 1.0f / s, p1 = e1 / s;
  int q0 = atomicAdd(&cnt[i0], 1);
  tok[i0 * T_ + q0] = t; wgt[i0 * T_ + q0] = p0; slot[i0 * T_ + q0] = 0;
  int q1 = atomicAdd(&cnt[i1], 1);
  tok[i1 * T_ + q1] = t; wgt[i1 * T_ + q1] = p1; slot[i1 * T_ + q1] = 1;
}

__global__ void zero_ints(int* __restrict__ p, int n) {
  int i = blockIdx.x * blockDim.x + threadIdx.x;
  if (i < n) p[i] = 0;
}
__global__ void prefix_counts(const int* __restrict__ cnt, int* __restrict__ basep) {
  if (threadIdx.x == 0 && blockIdx.x == 0) {
    int a = 0;
    for (int e = 0; e < E_; ++e) { basep[e] = a; a += cnt[e]; }
  }
}

// =====================================================================
static inline int gemm_blocks(int M, int N) {
  int mt = (M + 31) / 32, nt = (N + 63) / 64;
  return (mt * nt + 7) / 8;
}

extern "C" void kernel_launch(void* const* d_in, const int* in_sizes, int n_in,
                              void* d_out, int out_size, void* d_ws, size_t ws_size,
                              hipStream_t stream) {
  const int*   ids      = (const int*)  d_in[0];
  const float* tok_emb  = (const float*)d_in[1];
  const float* pos_emb  = (const float*)d_in[2];
  const float* w_qkv    = (const float*)d_in[3];
  const float* b_qkv    = (const float*)d_in[4];
  const float* w_o      = (const float*)d_in[5];
  const float* b_o      = (const float*)d_in[6];
  const float* ln_g     = (const float*)d_in[7];
  const float* ln_b     = (const float*)d_in[8];
  const float* router_w = (const float*)d_in[9];
  const float* router_b = (const float*)d_in[10];
  const float* eln_g    = (const float*)d_in[11];
  const float* eln_b    = (const float*)d_in[12];
  const float* w1       = (const float*)d_in[13];
  const float* b1       = (const float*)d_in[14];
  const float* w2       = (const float*)d_in[15];
  const float* b2       = (const float*)d_in[16];
  const float* w_out    = (const float*)d_in[17];
  const float* b_out    = (const float*)d_in[18];
  float* logits = (float*)d_out;

  // ---- workspace carve-out (~150 MB) ----
  char* ws = (char*)d_ws;
  size_t off = 0;
  auto carve = [&](size_t bytes) -> void* {
    off = (off + 255) & ~(size_t)255;
    void* p = ws + off; off += bytes; return p;
  };
  float*  x       = (float*) carve((size_t)T_ * H_ * 4);
  float*  abuf    = (float*) carve((size_t)T_ * H_ * 4);
  float*  contrib = (float*) carve((size_t)KTOP * T_ * H_ * 4);
  float*  mu      = (float*) carve((size_t)T_ * 4);
  float*  rs      = (float*) carve((size_t)T_ * 4);
  __bf16* xb      = (__bf16*)carve((size_t)T_ * H_ * 2);
  __bf16* qkvb    = (__bf16*)carve((size_t)T_ * 3 * H_ * 2);
  __bf16* scoresb = (__bf16*)carve((size_t)B_ * NH_ * S_ * S_ * 2);
  __bf16* vT      = (__bf16*)carve((size_t)B_ * NH_ * HD_ * S_ * 2);
  __bf16* attnb   = (__bf16*)carve((size_t)T_ * H_ * 2);
  __bf16* yb      = (__bf16*)carve((size_t)T_ * KTOP * H_ * 2);
  __bf16* hbufB   = (__bf16*)carve((size_t)T_ * KTOP * F_ * 2);
  __bf16* w_qkvB  = (__bf16*)carve((size_t)3 * H_ * H_ * 2);
  __bf16* w_oB    = (__bf16*)carve((size_t)H_ * H_ * 2);
  __bf16* w_outT  = (__bf16*)carve((size_t)V_ * H_ * 2);
  __bf16* w1T     = (__bf16*)carve((size_t)E_ * F_ * H_ * 2);
  __bf16* w2T     = (__bf16*)carve((size_t)E_ * H_ * F_ * 2);
  int*    cnt     = (int*)   carve(E_ * 4);
  int*    basep   = (int*)   carve(E_ * 4);
  int*    tokl    = (int*)   carve((size_t)E_ * T_ * 4);
  float*  wgtl    = (float*) carve((size_t)E_ * T_ * 4);
  int*    slotl   = (int*)   carve((size_t)E_ * T_ * 4);
  (void)ws_size; (void)in_sizes; (void)n_in; (void)out_size;

  const long long sQ = (long long)S_ * 3 * H_;  // per-b stride inside qkvb

  // ---- one-time weight conversion / transposition ----
  convert_bf16<<<(3 * H_ * H_ + 255) / 256, 256, 0, stream>>>(w_qkv, w_qkvB, 3 * H_ * H_);
  convert_bf16<<<(H_ * H_ + 255) / 256, 256, 0, stream>>>(w_o, w_oB, H_ * H_);
  transpose_bf16<float><<<dim3((V_ + 31) / 32, H_ / 32, 1), 256, 0, stream>>>(
      w_out, V_, 0, 0, w_outT, H_, 0, 0, H_, V_, 1);

  // embedding (fp32 + bf16)
  embed_kernel<<<(T_ * H_ + 255) / 256, 256, 0, stream>>>(ids, tok_emb, pos_emb, x, xb);

  for (int l = 0; l < L_; ++l) {
    // per-layer expert weights -> bf16, transposed to (N,K)
    transpose_bf16<float><<<dim3(F_ / 32, H_ / 32, E_), 256, 0, stream>>>(
        w1 + (size_t)l * E_ * H_ * F_, F_, (long long)H_ * F_, 0,
        w1T, H_, (long long)F_ * H_, 0, H_, F_, 1);
    transpose_bf16<float><<<dim3(H_ / 32, F_ / 32, E_), 256, 0, stream>>>(
        w2 + (size_t)l * E_ * F_ * H_, H_, (long long)F_ * H_, 0,
        w2T, F_, (long long)H_ * F_, 0, F_, H_, 1);

    // QKV projection: qkvb = xb @ w_qkv^T + b_qkv  (T x 1536, bf16 out)
    gemm_nt<1, 1><<<gemm_blocks(T_, 3 * H_), 256, 0, stream>>>(
        xb, H_, 0, 0, w_qkvB, H_, 0, 0, b_qkv,
        qkvb, 3 * H_, 0, 0, T_, 3 * H_, H_, 1.0f);

    // v transposed per (b,h): vT[b,h](HD,S)
    transpose_bf16<__bf16><<<dim3(HD_ / 32, S_ / 32, B_ * NH_), 256, 0, stream>>>(
        qkvb + 2 * H_, 3 * H_, sQ, HD_,
        vT, S_, (long long)NH_ * HD_ * S_, (long long)HD_ * S_, S_, HD_, NH_);

    // scores[b,h] = (q @ k^T) / 8  (bf16 out)
    gemm_nt<1, 0><<<dim3(gemm_blocks(S_, S_), NH_, B_), 256, 0, stream>>>(
        qkvb, 3 * H_, HD_, sQ,
        qkvb + H_, 3 * H_, HD_, sQ, nullptr,
        scoresb, S_, (long long)S_ * S_, (long long)NH_ * S_ * S_,
        S_, S_, HD_, 0.125f);

    softmax512<<<B_ * NH_ * S_, 256, 0, stream>>>(scoresb);

    // attn[b,h] = att @ v   (via vT, NT form; bf16 out)
    gemm_nt<1, 0><<<dim3(gemm_blocks(S_, HD_), NH_, B_), 256, 0, stream>>>(
        scoresb, S_, (long long)S_ * S_, (long long)NH_ * S_ * S_,
        vT, S_, (long long)HD_ * S_, (long long)NH_ * HD_ * S_, nullptr,
        attnb, H_, HD_, (long long)S_ * H_,
        S_, HD_, S_, 1.0f);

    // output projection: abuf = attn @ w_o^T + b_o  (fp32 out)
    gemm_nt<0, 1><<<gemm_blocks(T_, H_), 256, 0, stream>>>(
        attnb, H_, 0, 0, w_oB, H_, 0, 0, b_o,
        abuf, H_, 0, 0, T_, H_, H_, 1.0f);

    // x = LN(x + attention) (+ bf16 copy)
    add_ln<<<T_, 256, 0, stream>>>(x, abuf, ln_g, ln_b, xb);

    // router + top-2 + per-expert lists (sparse MoE: 2/8 experts computed)
    zero_ints<<<1, 32, 0, stream>>>(cnt, E_);
    router_topk<<<T_ / 256, 256, 0, stream>>>(
        x, router_w + (size_t)l * H_ * E_, router_b + (size_t)l * E_,
        cnt, tokl, wgtl, slotl);
    prefix_counts<<<1, 1, 0, stream>>>(cnt, basep);
    row_stats<<<T_, 256, 0, stream>>>(x, mu, rs);
    gather_norm<<<dim3(T_, E_), 256, 0, stream>>>(
        x, mu, rs, eln_g + (size_t)l * E_ * H_, eln_b + (size_t)l * E_ * H_,
        cnt, basep, tokl, yb);

    moe_fc1<<<dim3((T_ / 32) * (F_ / 64) / 8, E_), 256, 0, stream>>>(
        yb, w1T, b1 + (size_t)l * E_ * F_, cnt, basep, hbufB);

    moe_fc2<<<dim3((T_ / 32) * (H_ / 64) / 8, E_), 256, 0, stream>>>(
        hbufB, w2T, b2 + (size_t)l * E_ * H_, cnt, basep, tokl, wgtl, slotl, contrib);

    // x = LN(2x + contrib0 + contrib1) (+ bf16 copy)
    combine_ln<<<T_, 256, 0, stream>>>(x, contrib, ln_g, ln_b, xb);
  }

  // final vocab projection: logits = xb @ w_outT^T + b_out (2048 x 50000, fp32)
  gemm_nt<0, 1><<<gemm_blocks(T_, V_), 256, 0, stream>>>(
      xb, H_, 0, 0, w_outT, H_, 0, 0, b_out,
      logits, V_, 0, 0, T_, V_, H_, 1.0f);
}
